// LearnableTokenMergerV2_19318762898198
// MI455X (gfx1250) — matrix operand order repaired and checked by reading, not compiled, not run on hardware
//
#include <hip/hip_runtime.h>
#include <math.h>

// ---------------- CDNA5 WMMA types ----------------
typedef __attribute__((ext_vector_type(16))) __bf16 v16bf;
typedef __attribute__((ext_vector_type(8)))  float  v8f;
typedef __attribute__((ext_vector_type(4)))  float  v4f;

#define BM 128
#define BN 64
#define KT 64
#define LDA_U 33   // padded LDS row stride in dwords (each dword = 2 bf16 along K)
#define LDB_U 33

__device__ __forceinline__ unsigned int f2bf(float f) {
  unsigned int u = __float_as_uint(f);
  return (u + 0x7FFFu + ((u >> 16) & 1u)) >> 16;   // RNE bf16, in low 16 bits
}
__device__ __forceinline__ unsigned int packbf(float lo, float hi) {
  return f2bf(lo) | (f2bf(hi) << 16);
}

union FragBF { unsigned int u[8]; v16bf v; };

// ---------------------------------------------------------------------------
// Generic tiled GEMM:  C[row, col] = scale * sum_k Aop[row,k]*Bop[k,col] + bias[col]
//   Aop = transA ? A[k, row] : A[row, k]   (lda = leading dim of A as stored)
//   Bop = transB ? B[col, k] : B[k, col]   (ldb = leading dim of B as stored)
// Block tile 128x64, K-slab 64. 8 waves as 4(m) x 2(n); each wave owns 32x32
// (2x2 WMMA accumulators) -> 8 v_wmma_f32_16x16x32_bf16 per K-slab per wave.
// grid = (cols/BN, rows/BM, batch); batch index z -> (z/nInner, z%nInner) strides.
// ---------------------------------------------------------------------------
__global__ __launch_bounds__(256) void gemm_bf16_wmma(
    const float* __restrict__ Ag, const float* __restrict__ Bg,
    const float* __restrict__ bias, float* __restrict__ Cg,
    int K, int lda, int ldb, int ldc,
    int transA, int transB, int relu, float scale,
    int nInner,
    long long sAo, long long sAi, long long sBo, long long sBi,
    long long sCo, long long sCi)
{
  __shared__ __align__(16) unsigned int sA[BM * LDA_U];   // [row][k/2]
  __shared__ __align__(16) unsigned int sB[BN * LDB_U];   // [col][k/2]

  const int z  = blockIdx.z;
  const int zo = z / nInner, zi = z % nInner;
  Ag += zo * sAo + zi * sAi;
  Bg += zo * sBo + zi * sBi;
  Cg += zo * sCo + zi * sCi;

  const int row0 = blockIdx.y * BM;
  const int col0 = blockIdx.x * BN;
  const int tid  = threadIdx.x;
  const int lane = tid & 31;
  const int wave = tid >> 5;
  const int wm   = wave >> 1;   // 0..3 : 32-row band
  const int wn   = wave & 1;    // 0..1 : 32-col band

  v8f acc[2][2];
#pragma unroll
  for (int mt = 0; mt < 2; ++mt)
#pragma unroll
    for (int nt = 0; nt < 2; ++nt)
      acc[mt][nt] = (v8f){0.f, 0.f, 0.f, 0.f, 0.f, 0.f, 0.f, 0.f};

  for (int k0 = 0; k0 < K; k0 += KT) {
    // ---------------- stage A slab (BM x KT) ----------------
    if (!transA) {
      // thread: one row, 32 consecutive k (8 x float4 -> 16 packed dwords)
      const int r  = tid >> 1;                // 0..127
      const int kb = (tid & 1) * 32;          // 0 or 32
      const float* src = Ag + (size_t)(row0 + r) * lda + (k0 + kb);
      unsigned int* dst = &sA[r * LDA_U + (kb >> 1)];
#pragma unroll
      for (int vq = 0; vq < 8; ++vq) {
        v4f x = *(const v4f*)(src + 4 * vq);
        dst[2 * vq]     = packbf(x.x, x.y);
        dst[2 * vq + 1] = packbf(x.z, x.w);
      }
    } else {
      // thread: k-pair (2 rows of A), 16 consecutive i (4 x float4 per row)
      const int kp = tid >> 3;                // 0..31  -> k = 2*kp
      const int ib = (tid & 7) * 16;          // 0..112
      const float* s0 = Ag + (size_t)(k0 + 2 * kp) * lda + (row0 + ib);
      const float* s1 = s0 + lda;
#pragma unroll
      for (int vq = 0; vq < 4; ++vq) {
        v4f a0 = *(const v4f*)(s0 + 4 * vq);
        v4f a1 = *(const v4f*)(s1 + 4 * vq);
        sA[(ib + 4 * vq + 0) * LDA_U + kp] = packbf(a0.x, a1.x);
        sA[(ib + 4 * vq + 1) * LDA_U + kp] = packbf(a0.y, a1.y);
        sA[(ib + 4 * vq + 2) * LDA_U + kp] = packbf(a0.z, a1.z);
        sA[(ib + 4 * vq + 3) * LDA_U + kp] = packbf(a0.w, a1.w);
      }
    }
    // ---------------- stage B slab (BN x KT, stored [col][k/2]) ----------------
    if (transB) {
      // B = W[col][k] row-major: thread: one col, 16 consecutive k
      const int n  = tid >> 2;                // 0..63
      const int kb = (tid & 3) * 16;          // 0..48
      const float* src = Bg + (size_t)(col0 + n) * ldb + (k0 + kb);
      unsigned int* dst = &sB[n * LDB_U + (kb >> 1)];
#pragma unroll
      for (int vq = 0; vq < 4; ++vq) {
        v4f x = *(const v4f*)(src + 4 * vq);
        dst[2 * vq]     = packbf(x.x, x.y);
        dst[2 * vq + 1] = packbf(x.z, x.w);
      }
    } else {
      // B[k][col] row-major: thread: k-pair, 8 consecutive cols
      const int kp = tid >> 3;                // 0..31  -> k = 2*kp
      const int nb = (tid & 7) * 8;           // 0..56
      const float* s0 = Bg + (size_t)(k0 + 2 * kp) * ldb + (col0 + nb);
      const float* s1 = s0 + ldb;
#pragma unroll
      for (int vq = 0; vq < 2; ++vq) {
        v4f b0 = *(const v4f*)(s0 + 4 * vq);
        v4f b1 = *(const v4f*)(s1 + 4 * vq);
        sB[(nb + 4 * vq + 0) * LDB_U + kp] = packbf(b0.x, b1.x);
        sB[(nb + 4 * vq + 1) * LDB_U + kp] = packbf(b0.y, b1.y);
        sB[(nb + 4 * vq + 2) * LDB_U + kp] = packbf(b0.z, b1.z);
        sB[(nb + 4 * vq + 3) * LDB_U + kp] = packbf(b0.w, b1.w);
      }
    }
    __syncthreads();

    // ---------------- 2 K-steps of 32, 4 WMMAs each ----------------
#pragma unroll
    for (int ks = 0; ks < 2; ++ks) {          // ks*32 within slab
      FragBF fa[2], fb[2];
      const int kh2 = (lane >> 4) * 4 + ks * 16;   // A frag dword base along k/2
      const int kb2 = (lane >> 4) * 4 + ks * 16;   // B frag dword base along k/2
#pragma unroll
      for (int mt = 0; mt < 2; ++mt) {
        const int m = wm * 32 + mt * 16 + (lane & 15);
        const unsigned int* pa = &sA[m * LDA_U + kh2];
#pragma unroll
        for (int j = 0; j < 8; ++j)
          fa[mt].u[j] = pa[(j < 4) ? j : (8 + (j - 4))];
      }
#pragma unroll
      for (int nt = 0; nt < 2; ++nt) {
        const int n = wn * 32 + nt * 16 + (lane & 15);
        const unsigned int* pb = &sB[n * LDB_U + kb2];
#pragma unroll
        for (int j = 0; j < 8; ++j)
          fb[nt].u[j] = pb[(j < 4) ? j : (8 + (j - 4))];
      }
#pragma unroll
      for (int mt = 0; mt < 2; ++mt)
#pragma unroll
        for (int nt = 0; nt < 2; ++nt)
          acc[mt][nt] = __builtin_amdgcn_wmma_f32_16x16x32_bf16(
              false, fa[mt].v, false, fb[nt].v,
              (short)0, acc[mt][nt], false, false);
    }
    __syncthreads();
  }

  // ---- write C: VGPR r -> M=r (lanes 0-15) / M=r+8 (lanes 16-31), N=lane&15 ----
  const int mg = (lane >> 4) * 8;
#pragma unroll
  for (int nt = 0; nt < 2; ++nt) {
    const int col = col0 + wn * 32 + nt * 16 + (lane & 15);
    const float bval = bias ? bias[col] : 0.f;
#pragma unroll
    for (int mt = 0; mt < 2; ++mt) {
#pragma unroll
      for (int r = 0; r < 8; ++r) {
        int row = row0 + wm * 32 + mt * 16 + mg + r;
        float v = acc[mt][nt][r] * scale + bval;
        if (relu) v = fmaxf(v, 0.f);
        Cg[(size_t)row * ldc + col] = v;
      }
    }
  }
}

// --------- softmax over 64 elements, one wave32 per row (2 elems/lane) ---------
__global__ __launch_bounds__(256) void softmax64_kernel(float* __restrict__ x, int rows)
{
  int row = blockIdx.x * 8 + (threadIdx.x >> 5);
  if (row >= rows) return;
  int lane = threadIdx.x & 31;
  float* p = x + (size_t)row * 64;
  float a = p[lane], b = p[lane + 32];
  float m = fmaxf(a, b);
#pragma unroll
  for (int o = 16; o > 0; o >>= 1) m = fmaxf(m, __shfl_xor(m, o, 32));
  float ea = __expf(a - m), eb = __expf(b - m);
  float s = ea + eb;
#pragma unroll
  for (int o = 16; o > 0; o >>= 1) s += __shfl_xor(s, o, 32);
  float inv = 1.f / s;
  p[lane] = ea * inv;
  p[lane + 32] = eb * inv;
}

// --------- softmax over `len` elements, one 256-thread block per row ---------
__global__ __launch_bounds__(256) void softmax_row_block(float* __restrict__ x, int len)
{
  __shared__ float red[256];
  float* p = x + (size_t)blockIdx.x * len;
  const int tid = threadIdx.x;
  float m = -3.4e38f;
  for (int i = tid; i < len; i += 256) m = fmaxf(m, p[i]);
  red[tid] = m; __syncthreads();
  for (int s = 128; s > 0; s >>= 1) { if (tid < s) red[tid] = fmaxf(red[tid], red[tid + s]); __syncthreads(); }
  m = red[0]; __syncthreads();
  float sum = 0.f;
  for (int i = tid; i < len; i += 256) { float e = __expf(p[i] - m); p[i] = e; sum += e; }
  red[tid] = sum; __syncthreads();
  for (int s = 128; s > 0; s >>= 1) { if (tid < s) red[tid] += red[tid + s]; __syncthreads(); }
  float inv = 1.f / red[0];
  for (int i = tid; i < len; i += 256) p[i] *= inv;
}

// --------- importance logit: out[row] = dot(h[row,:K], w2) + b2 (wave/row) ---------
__global__ __launch_bounds__(256) void importance_kernel(
    const float* __restrict__ h, const float* __restrict__ w2,
    const float* __restrict__ b2, float* __restrict__ out, int rows, int K)
{
  int row = blockIdx.x * 8 + (threadIdx.x >> 5);
  if (row >= rows) return;
  int lane = threadIdx.x & 31;
  const float* p = h + (size_t)row * K;
  float s = 0.f;
  for (int k = lane; k < K; k += 32) s += p[k] * w2[k];
#pragma unroll
  for (int o = 16; o > 0; o >>= 1) s += __shfl_xor(s, o, 32);
  if (lane == 0) out[row] = s + b2[0];
}

// --------- exact top-k by rank counting (descending, index tiebreak) ---------
__global__ __launch_bounds__(256) void topk_kernel(
    const float* __restrict__ logits, int n, int m,
    float* __restrict__ idxF, int* __restrict__ idxI)
{
  __shared__ float s[2048];
  int b = blockIdx.x;
  const float* L = logits + (size_t)b * n;
  for (int i = threadIdx.x; i < n; i += 256) s[i] = L[i];
  __syncthreads();
  for (int i = threadIdx.x; i < n; i += 256) {
    float v = s[i];
    int cnt = 0;
    for (int j = 0; j < n; ++j) {
      float u = s[j];
      cnt += (u > v) || (u == v && j < i);
    }
    if (cnt < m) {
      idxI[(size_t)b * m + cnt] = i;
      idxF[(size_t)b * m + cnt] = (float)i;
    }
  }
}

// --------- gather rows: dst[b,m,:] = src[b, idx[b,m], :] ---------
__global__ __launch_bounds__(256) void gather_rows_kernel(
    const float* __restrict__ src, const int* __restrict__ idx,
    float* __restrict__ dst, int M, int D, int N)
{
  int row = blockIdx.x;             // b*M + m
  int b = row / M;
  int id = idx[row];
  const float* s = src + ((size_t)b * N + id) * D;
  float* d = dst + (size_t)row * D;
  for (int t = threadIdx.x; t < D; t += 256) d[t] = s[t];
}

// ---------------------------------------------------------------------------
static inline void launch_gemm(hipStream_t stream,
    const float* A, const float* Bmat, const float* bias, float* C,
    int rows, int cols, int K, int lda, int ldb, int ldc,
    int transA, int transB, int relu, float scale,
    int batch, int nInner,
    long long sAo, long long sAi, long long sBo, long long sBi,
    long long sCo, long long sCi)
{
  dim3 grid(cols / BN, rows / BM, batch);
  gemm_bf16_wmma<<<grid, dim3(256), 0, stream>>>(
      A, Bmat, bias, C, K, lda, ldb, ldc, transA, transB, relu, scale,
      nInner, sAo, sAi, sBo, sBi, sCo, sCi);
}

extern "C" void kernel_launch(void* const* d_in, const int* in_sizes, int n_in,
                              void* d_out, int out_size, void* d_ws, size_t ws_size,
                              hipStream_t stream) {
  (void)in_sizes; (void)n_in; (void)out_size; (void)ws_size;
  const int Bb = 8, N = 2048, NT = 64, Dv = 1024, Dt = 4096, Di = 512, H = 4, dh = 128, M = 1024;

  const float* vision = (const float*)d_in[0];
  const float* quest  = (const float*)d_in[1];
  const float* vp_w = (const float*)d_in[2];  const float* vp_b = (const float*)d_in[3];
  const float* tp_w = (const float*)d_in[4];  const float* tp_b = (const float*)d_in[5];
  const float* inq_w = (const float*)d_in[6]; const float* inq_b = (const float*)d_in[7];
  const float* ink_w = (const float*)d_in[8]; const float* ink_b = (const float*)d_in[9];
  const float* inv_w = (const float*)d_in[10]; const float* inv_b = (const float*)d_in[11];
  const float* outp_w = (const float*)d_in[12]; const float* outp_b = (const float*)d_in[13];
  const float* imp1_w = (const float*)d_in[14]; const float* imp1_b = (const float*)d_in[15];
  const float* imp2_w = (const float*)d_in[16]; const float* imp2_b = (const float*)d_in[17];
  const float* q_w = (const float*)d_in[18]; const float* q_b = (const float*)d_in[19];
  const float* k_w = (const float*)d_in[20]; const float* k_b = (const float*)d_in[21];
  const float* o_w = (const float*)d_in[22]; const float* o_b = (const float*)d_in[23];

  // ---- workspace carving (floats) ----
  float* ws = (float*)d_ws;
  size_t off = 0;
  float* bufV    = ws + off; off += (size_t)Bb * N * Di;      // V projection
  float* bufQt   = ws + off; off += (size_t)Bb * NT * Di;     // text projection
  float* bufq    = ws + off; off += (size_t)Bb * N * Di;      // q (later Q_merge)
  float* bufk    = ws + off; off += (size_t)Bb * NT * Di;
  float* bufv    = ws + off; off += (size_t)Bb * NT * Di;
  float* bufattn = ws + off; off += (size_t)Bb * H * N * NT;  // (B,H,N,T)
  float* bufctx  = ws + off; off += (size_t)Bb * N * Di;      // ctx (later merged_internal)
  float* bufatt  = ws + off; off += (size_t)Bb * N * Di;      // attended_V
  float* bufcc   = ws + off; off += (size_t)Bb * M * Di;      // cluster centers
  float* bufKm   = ws + off; off += (size_t)Bb * M * Di;      // K_merge
  int*   bufIdx  = (int*)(ws + off); off += (size_t)Bb * M;
  float* bufh    = bufV;    // h MLP hidden reuses V (V dead by then)
  float* bufQm   = bufq;    // Q_merge reuses q
  float* bufmi   = bufctx;  // merged_internal reuses ctx

  // ---- output regions (concatenated in reference return order) ----
  float* out    = (float*)d_out;
  float* out_mf  = out;                                  // (B,M,Dv)
  float* out_idx = out_mf + (size_t)Bb * M * Dv;         // (B,M)
  float* out_mw  = out_idx + (size_t)Bb * M;             // (B,N,M)
  float* out_lg  = out_mw + (size_t)Bb * N * M;          // (B,N)

  const float s_attn  = 1.0f / sqrtf((float)dh);
  const float s_merge = 1.0f / sqrtf((float)Di);

  // 1) V = vision @ vp_w^T + vp_b            (16384 x 512, K=1024)
  launch_gemm(stream, vision, vp_w, vp_b, bufV, Bb * N, Di, Dv, Dv, Dv, Di,
              0, 1, 0, 1.f, 1, 1, 0, 0, 0, 0, 0, 0);
  // 2) Qt = quest @ tp_w^T + tp_b            (512 x 512, K=4096)
  launch_gemm(stream, quest, tp_w, tp_b, bufQt, Bb * NT, Di, Dt, Dt, Dt, Di,
              0, 1, 0, 1.f, 1, 1, 0, 0, 0, 0, 0, 0);
  // 3-5) q/k/v projections
  launch_gemm(stream, bufV, inq_w, inq_b, bufq, Bb * N, Di, Di, Di, Di, Di,
              0, 1, 0, 1.f, 1, 1, 0, 0, 0, 0, 0, 0);
  launch_gemm(stream, bufQt, ink_w, ink_b, bufk, Bb * NT, Di, Di, Di, Di, Di,
              0, 1, 0, 1.f, 1, 1, 0, 0, 0, 0, 0, 0);
  launch_gemm(stream, bufQt, inv_w, inv_b, bufv, Bb * NT, Di, Di, Di, Di, Di,
              0, 1, 0, 1.f, 1, 1, 0, 0, 0, 0, 0, 0);
  // 6) scores[b,h] = q[b,:,h*dh:] @ k[b,:,h*dh:]^T / sqrt(dh)   (N x T, K=dh), batch=B*H
  launch_gemm(stream, bufq, bufk, nullptr, bufattn, N, NT, dh, Di, Di, NT,
              0, 1, 0, s_attn, Bb * H, H,
              (long long)N * Di, (long long)dh,
              (long long)NT * Di, (long long)dh,
              (long long)H * N * NT, (long long)N * NT);
  // 7) softmax over T=64
  softmax64_kernel<<<dim3((Bb * H * N) / 8), dim3(256), 0, stream>>>(bufattn, Bb * H * N);
  // 8) ctx[b,h] = attn[b,h] @ v[b,:,h*dh:]    (N x dh, K=T)
  launch_gemm(stream, bufattn, bufv, nullptr, bufctx, N, dh, NT, NT, Di, Di,
              0, 0, 0, 1.f, Bb * H, H,
              (long long)H * N * NT, (long long)N * NT,
              (long long)NT * Di, (long long)dh,
              (long long)N * Di, (long long)dh);
  // 9) attended_V = ctx @ outp_w^T + outp_b
  launch_gemm(stream, bufctx, outp_w, outp_b, bufatt, Bb * N, Di, Di, Di, Di, Di,
              0, 1, 0, 1.f, 1, 1, 0, 0, 0, 0, 0, 0);
  // 10) h = relu(attended_V @ imp1_w^T + b)   (16384 x 256, K=512)
  launch_gemm(stream, bufatt, imp1_w, imp1_b, bufh, Bb * N, Di / 2, Di, Di, Di, Di / 2,
              0, 1, 1, 1.f, 1, 1, 0, 0, 0, 0, 0, 0);
  // 11) importance_logits = h @ imp2_w^T + b  -> out_lg
  importance_kernel<<<dim3((Bb * N) / 8), dim3(256), 0, stream>>>(
      bufh, imp2_w, imp2_b, out_lg, Bb * N, Di / 2);
  // 12) deterministic top-k (descending, index tiebreak)
  topk_kernel<<<dim3(Bb), dim3(256), 0, stream>>>(out_lg, N, M, out_idx, bufIdx);
  // 13) cluster_centers gather
  gather_rows_kernel<<<dim3(Bb * M), dim3(256), 0, stream>>>(bufatt, bufIdx, bufcc, M, Di, N);
  // 14) Q_merge = attended_V @ q_w^T + q_b
  launch_gemm(stream, bufatt, q_w, q_b, bufQm, Bb * N, Di, Di, Di, Di, Di,
              0, 1, 0, 1.f, 1, 1, 0, 0, 0, 0, 0, 0);
  // 15) K_merge = cluster_centers @ k_w^T + k_b   (8192 x 512 flat)
  launch_gemm(stream, bufcc, k_w, k_b, bufKm, Bb * M, Di, Di, Di, Di, Di,
              0, 1, 0, 1.f, 1, 1, 0, 0, 0, 0, 0, 0);
  // 16) similarity[b] = Q_merge[b] @ K_merge[b]^T / sqrt(Di)  -> out_mw (N x M)
  launch_gemm(stream, bufQm, bufKm, nullptr, out_mw, N, M, Di, Di, Di, M,
              0, 1, 0, s_merge, Bb, 1,
              (long long)N * Di, 0,
              (long long)M * Di, 0,
              (long long)N * M, 0);
  // 17) merge_weights = softmax over M (in place in output)
  softmax_row_block<<<dim3(Bb * N), dim3(256), 0, stream>>>(out_mw, M);
  // 18) merged_internal[b] = mw[b]^T @ attended_V[b]   (M x Di, K=N) — transA path
  launch_gemm(stream, out_mw, bufatt, nullptr, bufmi, M, Di, N, M, Di, Di,
              1, 0, 0, 1.f, Bb, 1,
              (long long)N * M, 0,
              (long long)N * Di, 0,
              (long long)M * Di, 0);
  // 19) merged_features = merged_internal @ o_w^T + o_b   (8192 x 1024 flat) -> out_mf
  launch_gemm(stream, bufmi, o_w, o_b, out_mf, Bb * M, Dv, Di, Di, Di, Dv,
              0, 1, 0, 1.f, 1, 1, 0, 0, 0, 0, 0, 0);
}
